// Head_12876311953997
// MI455X (gfx1250) — matrix operand order repaired and checked by reading, not compiled, not run on hardware
//
#include <hip/hip_runtime.h>
#include <cstdint>

// ---------------------------------------------------------------------------
// CDNA5 (gfx1250) D-MPNN: bf16 WMMA GEMMs + fused GRU + graph gathers.
// Algebra: h[bgraph] @ Ur == (h @ Ur)[bgraph]  (NB=6x FLOP cut on the dominant
// matmul); loop-invariant hmess halves of the Wz/Wh concat-matmuls hoisted.
// Register blocking sized to avoid scratch spills: single GEMM uses
// 2 row x 4 col tiles (64 acc VGPRs); GRU dual-GEMM uses 1 row x 4 col x 2
// sets (64 acc VGPRs). B weight fragments are reused across row tiles.
// ---------------------------------------------------------------------------

typedef __bf16 bf16_t;
typedef __attribute__((ext_vector_type(16))) __bf16 v16bf;
typedef __attribute__((ext_vector_type(8)))  __bf16 v8bf;
typedef __attribute__((ext_vector_type(8)))  float  v8f;

#define HDIM 256

__device__ __forceinline__ float sigmoidf_(float x) {
    return 1.0f / (1.0f + __expf(-x));
}

__device__ __forceinline__ v8f wmma_bf16(v16bf a, v16bf b, v8f c) {
    // D = A(16x32 bf16) * B(32x16 bf16) + C(16x16 f32)
    return __builtin_amdgcn_wmma_f32_16x16x32_bf16(
        /*neg_a=*/false, a, /*neg_b=*/false, b,
        /*c_mod=*/(short)0, c, /*reuse_a=*/false, /*reuse_b=*/false);
}

// A fragment: 16x32 bf16 tile, row-major A with leading dim lda.
// lane<16 : row M=lane,    K = {kc+0..7, kc+16..23}
// lane>=16: row M=lane-16, K = {kc+8..15, kc+24..31}
__device__ __forceinline__ v16bf load_a_frag(const bf16_t* __restrict__ A, int lda,
                                             int r0, int kc, int lane) {
    int m  = r0 + (lane & 15);
    int kb = (lane < 16) ? 0 : 8;
    const bf16_t* p = A + (size_t)m * lda + kc + kb;
    v8bf lo = *(const v8bf*)p;
    v8bf hi = *(const v8bf*)(p + 16);
    v16bf a;
#pragma unroll
    for (int i = 0; i < 8; ++i) { a[i] = lo[i]; a[8 + i] = hi[i]; }
    return a;
}

// B fragment: 32x16 bf16 tile from column-major-stored weights Wt[256][ldw]
// (Wt[n*ldw + k] == W[k][n]).
// lane<16 : col N=lane,    K = {kc+0..7, kc+16..23}
// lane>=16: col N=lane-16, K = {kc+8..15, kc+24..31}
__device__ __forceinline__ v16bf load_b_frag(const bf16_t* __restrict__ Wt, int ldw,
                                             int n0, int kc, int lane) {
    int n  = n0 + (lane & 15);
    int kb = (lane < 16) ? 0 : 8;
    const bf16_t* p = Wt + (size_t)n * ldw + kc + kb;
    v8bf lo = *(const v8bf*)p;
    v8bf hi = *(const v8bf*)(p + 16);
    v16bf b;
#pragma unroll
    for (int i = 0; i < 8; ++i) { b[i] = lo[i]; b[8 + i] = hi[i]; }
    return b;
}

// ---------------------------------------------------------------------------
// GEMM: C[M,256] = A[M,K]_bf16 @ Wt[K,256]_bf16 (+ bias; RELU/mask optional).
// Block = 256 threads = 8 wave32; block tile 64x256; wave tile 32x64
// (2 row tiles x 4 col tiles = 64 acc VGPRs, spill-free). B fragments are
// loaded once per K-chunk and reused across both row tiles.
// ---------------------------------------------------------------------------
template <int K, int RELU_MASK>
__global__ __launch_bounds__(256)
void gemm_kernel(const bf16_t* __restrict__ A,
                 const bf16_t* __restrict__ Wt,
                 const float* __restrict__ bias,
                 float* __restrict__ C, int M) {
    const int lane  = threadIdx.x & 31;
    const int w     = threadIdx.x >> 5;
    const int rbase = blockIdx.x * 64 + (w >> 2) * 32;
    const int c0    = (w & 3) * 64;

    int rs[2];
#pragma unroll
    for (int i = 0; i < 2; ++i)
        rs[i] = (rbase + i * 16 < M) ? (rbase + i * 16) : 0;  // clamp, skip store

    v8f acc[2][4] = {};
#pragma unroll
    for (int kc = 0; kc < K; kc += 32) {
        v16bf b[4];
#pragma unroll
        for (int t = 0; t < 4; ++t)
            b[t] = load_b_frag(Wt, K, c0 + t * 16, kc, lane);
#pragma unroll
        for (int i = 0; i < 2; ++i) {
            v16bf a = load_a_frag(A, K, rs[i], kc, lane);
#pragma unroll
            for (int t = 0; t < 4; ++t)
                acc[i][t] = wmma_bf16(a, b[t], acc[i][t]);
        }
    }

    const int nn = (lane & 15);
#pragma unroll
    for (int i = 0; i < 2; ++i) {
        const int r0 = rbase + i * 16;
        if (r0 >= M) continue;                     // wave-uniform guard
        const int mb = r0 + ((lane >= 16) ? 8 : 0);
#pragma unroll
        for (int t = 0; t < 4; ++t) {
            const int n  = c0 + t * 16 + nn;
            const float bv = bias ? bias[n] : 0.0f;
#pragma unroll
            for (int j = 0; j < 8; ++j) {
                const int m = mb + j;
                float v = acc[i][t][j] + bv;
                if (RELU_MASK) {
                    v = (v > 0.0f) ? v : 0.0f;
                    if (m == 0) v = 0.0f;          // n_mask: node 0 is padding
                }
                C[(size_t)m * HDIM + n] = v;
            }
        }
    }
}

// ---------------------------------------------------------------------------
// Fused GRU update: dual GEMM  Cz = sum_h@Wz2, Ch = sum_gh@Wh2,  then
//   z = sigmoid(mz + Cz); pre_h = tanh(mh + Ch);
//   h_new = (1-z)*sum_h + z*pre_h ; row 0 masked (padding message).
// Block tile 32x256; wave tile 16x64 per matrix (1 row x 4 col x 2 sets
// = 64 acc VGPRs, spill-free).
// ---------------------------------------------------------------------------
__global__ __launch_bounds__(256)
void gru_update_kernel(const bf16_t* __restrict__ Ah,   // sum_h  bf16 [E,256]
                       const bf16_t* __restrict__ Ag,   // sum_gh bf16 [E,256]
                       const bf16_t* __restrict__ Wz2t, // [256,256]
                       const bf16_t* __restrict__ Wh2t, // [256,256]
                       const float* __restrict__ mz,    // hmess@Wz1+bz
                       const float* __restrict__ mh,    // hmess@Wh1+bh
                       const float* __restrict__ sumh,  // f32 [E,256]
                       float* __restrict__ h,
                       bf16_t* __restrict__ hbf, int M) {
    const int lane = threadIdx.x & 31;
    const int w    = threadIdx.x >> 5;
    const int r0   = blockIdx.x * 32 + (w >> 2) * 16;
    const int c0   = (w & 3) * 64;
    if (r0 >= M) return;

    v8f accz[4] = {};
    v8f acch[4] = {};
#pragma unroll
    for (int kc = 0; kc < HDIM; kc += 32) {
        v16bf a1 = load_a_frag(Ah, HDIM, r0, kc, lane);
        v16bf a2 = load_a_frag(Ag, HDIM, r0, kc, lane);
#pragma unroll
        for (int t = 0; t < 4; ++t) {
            v16bf b1 = load_b_frag(Wz2t, HDIM, c0 + t * 16, kc, lane);
            accz[t] = wmma_bf16(a1, b1, accz[t]);
            v16bf b2 = load_b_frag(Wh2t, HDIM, c0 + t * 16, kc, lane);
            acch[t] = wmma_bf16(a2, b2, acch[t]);
        }
    }

    const int mb = r0 + ((lane >= 16) ? 8 : 0);
    const int nn = (lane & 15);
#pragma unroll
    for (int t = 0; t < 4; ++t) {
        const int n = c0 + t * 16 + nn;
#pragma unroll
        for (int j = 0; j < 8; ++j) {
            const int m = mb + j;
            const size_t idx = (size_t)m * HDIM + n;
            float z  = sigmoidf_(mz[idx] + accz[t][j]);
            float ph = tanhf(mh[idx] + acch[t][j]);
            float hn = (1.0f - z) * sumh[idx] + z * ph;
            if (m == 0) hn = 0.0f;                 // e_mask: message 0 is padding
            h[idx]   = hn;
            hbf[idx] = (bf16_t)hn;
        }
    }
}

// ---------------------------------------------------------------------------
// Neighbor gather: per edge e, over NB=6 inbound messages b=bgraph[e,n]:
//   sum_h  = sum h[b]
//   sum_gh = sum sigmoid(r1[e] + hu[b] + bur) * h[b]    (hu = h@Ur, pre-GEMMed)
// ---------------------------------------------------------------------------
__global__ __launch_bounds__(256)
void gather_kernel(const float* __restrict__ h,
                   const float* __restrict__ hu,
                   const float* __restrict__ r1,
                   const int* __restrict__ bgraph,
                   const float* __restrict__ bur,
                   float* __restrict__ sumh,
                   bf16_t* __restrict__ sumh_bf,
                   bf16_t* __restrict__ sumgh_bf) {
    const int e = blockIdx.x;
    const int j = threadIdx.x;          // 256 threads = H columns
    const float r1v = r1[(size_t)e * HDIM + j];
    const float bv  = bur[j];
    float sh = 0.0f, sg = 0.0f;
#pragma unroll
    for (int n = 0; n < 6; ++n) {
        const int b = bgraph[e * 6 + n];
        const float hv = h[(size_t)b * HDIM + j];
        const float ru = sigmoidf_(r1v + hu[(size_t)b * HDIM + j] + bv);
        sh += hv;
        sg += ru * hv;
    }
    const size_t idx = (size_t)e * HDIM + j;
    sumh[idx]     = sh;
    sumh_bf[idx]  = (bf16_t)sh;
    sumgh_bf[idx] = (bf16_t)sg;
}

// hmess[e, 0:98]=fnode[src], [98:104]=bond feats, [104:128]=0 (K padding), bf16.
__global__ __launch_bounds__(128)
void hmess_kernel(const float* __restrict__ fnode,
                  const int* __restrict__ fmess_idx,
                  const float* __restrict__ fmess_feat,
                  bf16_t* __restrict__ hmess) {
    const int e = blockIdx.x;
    const int j = threadIdx.x;          // 128 threads
    const int src = fmess_idx[e * 2];
    float v = 0.0f;
    if (j < 98)       v = fnode[(size_t)src * 98 + j];
    else if (j < 104) v = fmess_feat[e * 6 + (j - 98)];
    hmess[(size_t)e * 128 + j] = (bf16_t)v;
}

// Readout input: Ao[v, 0:98]=fnode[v], [98:354]=sum_{agraph} h, [354:384]=0.
__global__ __launch_bounds__(384)
void ao_kernel(const float* __restrict__ fnode,
               const int* __restrict__ agraph,
               const float* __restrict__ h,
               bf16_t* __restrict__ Ao) {
    const int v = blockIdx.x;
    const int j = threadIdx.x;          // 384 threads
    float x = 0.0f;
    if (j < 98) {
        x = fnode[(size_t)v * 98 + j];
    } else if (j < 354) {
        const int c = j - 98;
#pragma unroll
        for (int n = 0; n < 6; ++n) {
            const int a = agraph[v * 6 + n];
            x += h[(size_t)a * HDIM + c];
        }
    }
    Ao[(size_t)v * 384 + j] = (bf16_t)x;
}

// Transpose + zero-pad weights into WMMA-friendly bf16 [256, Kpad] layout:
// dst[n*Kpad + k] = (k < Ksrc) ? src[k*256 + n] : 0.
__global__ __launch_bounds__(384)
void wprep_kernel(const float* __restrict__ src,
                  bf16_t* __restrict__ dst, int Ksrc, int Kpad) {
    const int n  = blockIdx.x;          // 256 output columns
    const int kk = threadIdx.x;         // Kpad threads
    const float v = (kk < Ksrc) ? src[(size_t)kk * HDIM + n] : 0.0f;
    dst[(size_t)n * Kpad + kk] = (bf16_t)v;
}

// ---------------------------------------------------------------------------
extern "C" void kernel_launch(void* const* d_in, const int* in_sizes, int n_in,
                              void* d_out, int out_size, void* d_ws, size_t ws_size,
                              hipStream_t stream) {
    (void)in_sizes; (void)n_in; (void)out_size; (void)ws_size;
    constexpr int N = 50000, E = 100000, H = HDIM, DEPTH = 6;

    const float* fnode      = (const float*)d_in[0];
    const int*   fmess_idx  = (const int*)  d_in[1];
    const float* fmess_feat = (const float*)d_in[2];
    const int*   agraph     = (const int*)  d_in[3];
    const int*   bgraph     = (const int*)  d_in[4];
    const float* Wz         = (const float*)d_in[5];
    const float* bz         = (const float*)d_in[6];
    const float* Wr         = (const float*)d_in[7];
    const float* Ur         = (const float*)d_in[8];
    const float* bur        = (const float*)d_in[9];
    const float* Wh         = (const float*)d_in[10];
    const float* bh         = (const float*)d_in[11];
    const float* Wo         = (const float*)d_in[12];
    const float* bo         = (const float*)d_in[13];
    float* out = (float*)d_out;

    // Workspace carve-up (256B aligned).
    char* p = (char*)d_ws;
    auto take = [&](size_t bytes) -> char* {
        char* r = p;
        p += (bytes + 255) & ~(size_t)255;
        return r;
    };
    bf16_t* hmess    = (bf16_t*)take((size_t)E * 128 * sizeof(bf16_t));
    float*  r1       = (float*) take((size_t)E * H * sizeof(float));
    float*  mz       = (float*) take((size_t)E * H * sizeof(float));
    float*  mh       = (float*) take((size_t)E * H * sizeof(float));
    float*  h        = (float*) take((size_t)E * H * sizeof(float));
    bf16_t* hbf      = (bf16_t*)take((size_t)E * H * sizeof(bf16_t));
    float*  hu       = (float*) take((size_t)E * H * sizeof(float));
    float*  sumh     = (float*) take((size_t)E * H * sizeof(float));
    bf16_t* sumh_bf  = (bf16_t*)take((size_t)E * H * sizeof(bf16_t));
    bf16_t* sumgh_bf = (bf16_t*)take((size_t)E * H * sizeof(bf16_t));
    bf16_t* Ao       = (bf16_t*)take((size_t)N * 384 * sizeof(bf16_t));
    bf16_t* Wrt      = (bf16_t*)take((size_t)H * 128 * sizeof(bf16_t));
    bf16_t* Wz1t     = (bf16_t*)take((size_t)H * 128 * sizeof(bf16_t));
    bf16_t* Wz2t     = (bf16_t*)take((size_t)H * 256 * sizeof(bf16_t));
    bf16_t* Wh1t     = (bf16_t*)take((size_t)H * 128 * sizeof(bf16_t));
    bf16_t* Wh2t     = (bf16_t*)take((size_t)H * 256 * sizeof(bf16_t));
    bf16_t* Urt      = (bf16_t*)take((size_t)H * 256 * sizeof(bf16_t));
    bf16_t* Wot      = (bf16_t*)take((size_t)H * 384 * sizeof(bf16_t));

    // Embed messages once (shared by all 3 encoders).
    hmess_kernel<<<E, 128, 0, stream>>>(fnode, fmess_idx, fmess_feat, hmess);

    const int gE  = (E + 63) / 64;      // single-GEMM grid over E rows
    const int gN  = (N + 63) / 64;      // single-GEMM grid over N rows
    const int gGE = (E + 31) / 32;      // GRU dual-GEMM grid over E rows

    for (int k = 0; k < 3; ++k) {
        const float* Wzk = Wz + (size_t)k * 360 * H;
        const float* Whk = Wh + (size_t)k * 360 * H;
        // Transpose/pad this encoder's weights to bf16 [256, Kpad].
        wprep_kernel<<<H, 128, 0, stream>>>(Wr + (size_t)k * 104 * H, Wrt, 104, 128);
        wprep_kernel<<<H, 128, 0, stream>>>(Wzk,                Wz1t, 104, 128);
        wprep_kernel<<<H, 256, 0, stream>>>(Wzk + 104 * H,      Wz2t, 256, 256);
        wprep_kernel<<<H, 128, 0, stream>>>(Whk,                Wh1t, 104, 128);
        wprep_kernel<<<H, 256, 0, stream>>>(Whk + 104 * H,      Wh2t, 256, 256);
        wprep_kernel<<<H, 256, 0, stream>>>(Ur + (size_t)k * H * H, Urt, 256, 256);
        wprep_kernel<<<H, 384, 0, stream>>>(Wo + (size_t)k * 354 * H, Wot, 354, 384);

        // Loop-invariant precomputes.
        gemm_kernel<128, 0><<<gE, 256, 0, stream>>>(hmess, Wrt,  nullptr,    r1, E);
        gemm_kernel<128, 0><<<gE, 256, 0, stream>>>(hmess, Wz1t, bz + k * H, mz, E);
        gemm_kernel<128, 0><<<gE, 256, 0, stream>>>(hmess, Wh1t, bh + k * H, mh, E);

        hipMemsetAsync(h,   0, (size_t)E * H * sizeof(float),  stream);
        hipMemsetAsync(hbf, 0, (size_t)E * H * sizeof(bf16_t), stream);

        for (int d = 0; d < DEPTH; ++d) {
            // hu = h @ Ur  (so h[bgraph]@Ur becomes a gather of hu)
            gemm_kernel<256, 0><<<gE, 256, 0, stream>>>(hbf, Urt, nullptr, hu, E);
            gather_kernel<<<E, 256, 0, stream>>>(h, hu, r1, bgraph, bur + k * H,
                                                 sumh, sumh_bf, sumgh_bf);
            gru_update_kernel<<<gGE, 256, 0, stream>>>(sumh_bf, sumgh_bf, Wz2t, Wh2t,
                                                       mz, mh, sumh, h, hbf, E);
        }

        // Node readout: out = relu([fnode | nei] @ Wo + bo), node 0 masked.
        ao_kernel<<<N, 384, 0, stream>>>(fnode, agraph, h, Ao);
        gemm_kernel<384, 1><<<gN, 256, 0, stream>>>(Ao, Wot, bo + k * H,
                                                    out + (size_t)k * N * H, N);
    }
}
